// MultiSetInversionModel_32272384262633
// MI455X (gfx1250) — matrix operand order repaired and checked by reading, not compiled, not run on hardware
//
#include <hip/hip_runtime.h>

// Problem dims (fixed by reference)
#define B_     256
#define E_     1024
#define V_     50000
#define G_     4096
#define STEPS_ 32

typedef __bf16 bf16_t;
typedef __attribute__((ext_vector_type(16))) __bf16 v16bf;
typedef __attribute__((ext_vector_type(8)))  __bf16 v8bf;
typedef __attribute__((ext_vector_type(8)))  float  v8f;
typedef int v4i_ __attribute__((vector_size(16)));   // matches builtin param type

#define AS1 __attribute__((address_space(1)))
#define AS3 __attribute__((address_space(3)))

// Async global->LDS staging path (gfx1250): guarded so either toolchain compiles.
#if defined(__has_builtin)
#if __has_builtin(__builtin_amdgcn_global_load_async_to_lds_b128)
#define USE_ASYNC 1
#endif
#endif
#ifndef USE_ASYNC
#define USE_ASYNC 0
#endif

__device__ __forceinline__ bf16_t f2bf(float f) {
  union { float f; unsigned u; } a; a.f = f;
  unsigned r = a.u + 0x7FFFu + ((a.u >> 16) & 1u);   // round-to-nearest-even
  union { unsigned short s; bf16_t b; } o; o.s = (unsigned short)(r >> 16);
  return o.b;
}
__device__ __forceinline__ float sigf(float x) { return 1.f / (1.f + __expf(-x)); }

// LDS A-slice row stride in halves: 32 data + 8 pad -> conflict-free 16B ds reads
#define APAD 40

// ---------------------------------------------------------------------------
// WMMA GEMM: out[M=256, N] = A1[M,K]*W1[N,K]^T (+ A2[M,K]*W2[N,K]^T) + bias
// bf16 in, fp32 accumulate (v_wmma_f32_16x16x32_bf16).
// Block = 128 threads = 4 waves. Wave -> one 16-col tile; block -> 64 rows.
// A slice (64x32) staged to LDS via async copy, double buffered; B fragment
// double buffered in registers. grid = (ceil(NT/4), 4).
// ---------------------------------------------------------------------------
template <bool DUAL>
__global__ __launch_bounds__(128) void gemm_wmma(
    const bf16_t* __restrict__ A1, const bf16_t* __restrict__ W1,
    const bf16_t* __restrict__ A2, const bf16_t* __restrict__ W2,
    const float* __restrict__ bias,
    float* __restrict__ outF, bf16_t* __restrict__ outB,
    int N, int K, int NT)
{
  __shared__ __align__(16) bf16_t sA1[2][64 * APAD];
  __shared__ __align__(16) bf16_t sA2[2][64 * APAD];

  const int wave  = threadIdx.x >> 5;
  const int lane  = threadIdx.x & 31;
  const int ntile = blockIdx.x * 4 + wave;
  const bool active = ntile < NT;
  const int nt    = active ? ntile : (NT - 1);   // clamp: wave stays for barriers/WMMA
  const int col   = nt * 16 + (lane & 15);
  const int mbase = blockIdx.y * 64;
  const int kloA  = (lane >> 4) * 16;  // A frag: 16 contiguous K per lane-half
  const int kloB  = (lane >> 4) * 16;  // B frag: 16 contiguous K per lane-half
  const int rA    = lane & 15;
  const int rowL  = threadIdx.x >> 1;        // staging: row within 64-row slice
  const int seg   = (threadIdx.x & 1) * 16;  // staging: which 16-half segment

  const size_t wrow = (size_t)col * K;
  const int nk = K >> 5;
  v8f acc[4] = {};

  auto stage = [&](int buf, int kc) {
    const int k0 = kc << 5;
    const bf16_t* s1 = A1 + (size_t)(mbase + rowL) * K + (size_t)(k0 + seg);
    bf16_t* d1 = &sA1[buf][rowL * APAD + seg];
#if USE_ASYNC
    __builtin_amdgcn_global_load_async_to_lds_b128((AS1 v4i_*)s1, (AS3 v4i_*)d1, 0, 0);
    __builtin_amdgcn_global_load_async_to_lds_b128((AS1 v4i_*)s1, (AS3 v4i_*)d1, 16, 0);
#else
    *(v8bf*)d1       = *(const v8bf*)s1;
    *(v8bf*)(d1 + 8) = *(const v8bf*)(s1 + 8);
#endif
    if (DUAL) {
      const bf16_t* s2 = A2 + (size_t)(mbase + rowL) * K + (size_t)(k0 + seg);
      bf16_t* d2 = &sA2[buf][rowL * APAD + seg];
#if USE_ASYNC
      __builtin_amdgcn_global_load_async_to_lds_b128((AS1 v4i_*)s2, (AS3 v4i_*)d2, 0, 0);
      __builtin_amdgcn_global_load_async_to_lds_b128((AS1 v4i_*)s2, (AS3 v4i_*)d2, 16, 0);
#else
      *(v8bf*)d2       = *(const v8bf*)s2;
      *(v8bf*)(d2 + 8) = *(const v8bf*)(s2 + 8);
#endif
    }
  };

  // Prologue: stage chunk 0, load B fragment for chunk 0
  stage(0, 0);
  v16bf b1 = *(const v16bf*)(W1 + wrow + (size_t)kloB);
  v16bf b2;
  if (DUAL) b2 = *(const v16bf*)(W2 + wrow + (size_t)kloB);

  for (int kc = 0; kc < nk; ++kc) {
    const int cur = kc & 1;
    const bool more = (kc + 1) < nk;
    if (more) stage(1 - cur, kc + 1);            // async copies for next chunk
#if USE_ASYNC
    if (more) {
      // leave only the just-issued stage outstanding -> current stage complete
      if (DUAL) asm volatile("s_wait_asynccnt 4" ::: "memory");
      else      asm volatile("s_wait_asynccnt 2" ::: "memory");
    } else {
      asm volatile("s_wait_asynccnt 0" ::: "memory");
    }
#endif
    __syncthreads();                             // all waves' staging visible

    // prefetch next B fragments into registers (overlaps with WMMAs below)
    v16bf b1n, b2n;
    if (more) {
      b1n = *(const v16bf*)(W1 + wrow + (size_t)(((kc + 1) << 5) + kloB));
      if (DUAL) b2n = *(const v16bf*)(W2 + wrow + (size_t)(((kc + 1) << 5) + kloB));
    }

#pragma unroll
    for (int m = 0; m < 4; ++m) {
      const bf16_t* ap = &sA1[cur][(m * 16 + rA) * APAD + kloA];
      union { v16bf v; v8bf p[2]; } a;
      a.p[0] = *(const v8bf*)ap;
      a.p[1] = *(const v8bf*)(ap + 8);
      acc[m] = __builtin_amdgcn_wmma_f32_16x16x32_bf16(
          false, a.v, false, b1, (short)0, acc[m], false, false);
    }
    if (DUAL) {
#pragma unroll
      for (int m = 0; m < 4; ++m) {
        const bf16_t* ap = &sA2[cur][(m * 16 + rA) * APAD + kloA];
        union { v16bf v; v8bf p[2]; } a;
        a.p[0] = *(const v8bf*)ap;
        a.p[1] = *(const v8bf*)(ap + 8);
        acc[m] = __builtin_amdgcn_wmma_f32_16x16x32_bf16(
            false, a.v, false, b2, (short)0, acc[m], false, false);
      }
    }
    __syncthreads();                             // protect buffer re-stage
    if (more) { b1 = b1n; if (DUAL) b2 = b2n; }
  }

  if (active) {
    const float bv = bias ? bias[col] : 0.f;
    const int r0 = (lane >> 4) * 8;              // C/D: VGPR r holds M=r (+8 upper lanes)
#pragma unroll
    for (int m = 0; m < 4; ++m) {
#pragma unroll
      for (int r = 0; r < 8; ++r) {
        const int row = mbase + m * 16 + r0 + r;
        const float v = acc[m][r] + bv;
        if (outF) outF[(size_t)row * N + col] = v;
        if (outB) outB[(size_t)row * N + col] = f2bf(v);
      }
    }
  }
}

// ---------------------------------------------------------------------------
// Small utility kernels
// ---------------------------------------------------------------------------
__global__ __launch_bounds__(256) void k_f2bf(const float* __restrict__ in,
                                              bf16_t* __restrict__ out, size_t n) {
  size_t i = (size_t)blockIdx.x * 256 + threadIdx.x;
  if (i < n) out[i] = f2bf(in[i]);
}
__global__ __launch_bounds__(256) void k_zero32(unsigned* __restrict__ p, size_t n) {
  size_t i = (size_t)blockIdx.x * 256 + threadIdx.x;
  if (i < n) p[i] = 0u;
}
__global__ __launch_bounds__(256) void k_copy32(const unsigned* __restrict__ s,
                                                unsigned* __restrict__ d, size_t n) {
  size_t i = (size_t)blockIdx.x * 256 + threadIdx.x;
  if (i < n) d[i] = s[i];
}
__global__ __launch_bounds__(256) void k_addbias(const float* __restrict__ a,
                                                 const float* __restrict__ b,
                                                 float* __restrict__ o, int n) {
  int i = blockIdx.x * 256 + threadIdx.x;
  if (i < n) o[i] = a[i] + b[i];
}

// LSTM cell elementwise: gates [B,4E] in torch order i,f,g,o
__global__ __launch_bounds__(256) void k_lstm(const float* __restrict__ gates,
                                              float* __restrict__ c,
                                              bf16_t* __restrict__ h_bf) {
  int idx = blockIdx.x * 256 + threadIdx.x;       // exactly B_*E_ threads
  int b = idx >> 10, e = idx & 1023;
  const float* g = gates + (size_t)b * G_;
  float ig = sigf(g[e]);
  float fg = sigf(g[E_ + e]);
  float gg = tanhf(g[2 * E_ + e]);
  float og = sigf(g[3 * E_ + e]);
  float cn = fg * c[idx] + ig * gg;
  c[idx] = cn;
  h_bf[idx] = f2bf(og * tanhf(cn));
}

// One block per batch row: fused argmax + online logsumexp + masked logit sum.
// Fix-up removes the predicted token from labels, sets pred, accumulates loss.
__global__ __launch_bounds__(256) void k_argmax_loss(
    const float* __restrict__ logits, unsigned char* __restrict__ labels,
    float* __restrict__ pred, int* __restrict__ yt,
    float* __restrict__ row_loss, int V)
{
  const int b = blockIdx.x, t = threadIdx.x;
  const float* lr = logits + (size_t)b * V;
  unsigned char* lb = labels + (size_t)b * V;

  float m = -1e30f, s = 0.f, bestv = -1e30f, msum = 0.f, cnt = 0.f;
  int besti = V;
  for (int v = t; v < V; v += 256) {
    float x = lr[v];
    if (x > m) { s = s * __expf(m - x) + 1.f; m = x; }
    else       { s += __expf(x - m); }
    if (x > bestv) { bestv = x; besti = v; }
    if (lb[v]) { msum += x; cnt += 1.f; }
  }

  __shared__ float sm[256], ss[256], sbv[256], sms[256], scn[256];
  __shared__ int   sbi[256];
  sm[t] = m; ss[t] = s; sbv[t] = bestv; sbi[t] = besti; sms[t] = msum; scn[t] = cnt;
  __syncthreads();
  for (int o = 128; o > 0; o >>= 1) {
    if (t < o) {
      float m2 = sm[t + o], s2 = ss[t + o];
      if (m2 > sm[t]) { ss[t] = ss[t] * __expf(sm[t] - m2) + s2; sm[t] = m2; }
      else            { ss[t] += s2 * __expf(m2 - sm[t]); }
      float bv2 = sbv[t + o]; int bi2 = sbi[t + o];
      if (bv2 > sbv[t] || (bv2 == sbv[t] && bi2 < sbi[t])) { sbv[t] = bv2; sbi[t] = bi2; }
      sms[t] += sms[t + o]; scn[t] += scn[t + o];
    }
    __syncthreads();
  }
  if (t == 0) {
    int y = sbi[0];
    float lse = sm[0] + __logf(ss[0]);
    float msumT = sms[0], cntT = scn[0];
    if (lb[y]) { msumT -= lr[y]; cntT -= 1.f; lb[y] = 0; }  // mask uses labels AFTER removing yt
    pred[(size_t)b * V + y] = 1.0f;
    yt[b] = y;
    row_loss[b] += cntT * lse - msumT;   // sum_v mask*(lse - logit)
  }
}

__global__ __launch_bounds__(256) void k_gather(const float* __restrict__ emb,
                                                const int* __restrict__ yt,
                                                bf16_t* __restrict__ x_bf) {
  int b = blockIdx.x;
  int y = yt[b];
  const float* src = emb + (size_t)y * E_;
  bf16_t* dst = x_bf + (size_t)b * E_;
  for (int e = threadIdx.x; e < E_; e += 256) dst[e] = f2bf(src[e]);
}

__global__ __launch_bounds__(256) void k_finalize(const float* __restrict__ row_loss,
                                                  float* __restrict__ out) {
  __shared__ float sh[256];
  int t = threadIdx.x;
  sh[t] = row_loss[t];
  __syncthreads();
  for (int o = 128; o > 0; o >>= 1) { if (t < o) sh[t] += sh[t + o]; __syncthreads(); }
  if (t == 0) out[0] = sh[0] * (1.0f / ((float)STEPS_ * (float)B_ * (float)V_));
}

// ---------------------------------------------------------------------------
extern "C" void kernel_launch(void* const* d_in, const int* in_sizes, int n_in,
                              void* d_out, int out_size, void* d_ws, size_t ws_size,
                              hipStream_t stream) {
  const float*         inputs = (const float*)d_in[0];
  const unsigned char* labels = (const unsigned char*)d_in[1];  // jnp.bool_ -> 1 byte/elem
  const float*         fc1_w  = (const float*)d_in[2];
  const float*         fc1_b  = (const float*)d_in[3];
  const float*         w_ih   = (const float*)d_in[4];
  const float*         w_hh   = (const float*)d_in[5];
  const float*         b_ih   = (const float*)d_in[6];
  const float*         b_hh   = (const float*)d_in[7];
  const float*         fc2_w  = (const float*)d_in[8];
  const float*         fc2_b  = (const float*)d_in[9];
  const float*         emb    = (const float*)d_in[10];

  // Workspace layout
  size_t off = 0;
  auto alloc = [&](size_t bytes) -> void* {
    void* p = (char*)d_ws + off;
    off += (bytes + 255) & ~(size_t)255;
    return p;
  };
  bf16_t* fc2w_bf = (bf16_t*)alloc((size_t)V_ * E_ * sizeof(bf16_t)); // 102.4 MB -> L2 resident
  bf16_t* wih_bf  = (bf16_t*)alloc((size_t)G_ * E_ * sizeof(bf16_t));
  bf16_t* whh_bf  = (bf16_t*)alloc((size_t)G_ * E_ * sizeof(bf16_t));
  bf16_t* fc1w_bf = (bf16_t*)alloc((size_t)E_ * E_ * sizeof(bf16_t));
  bf16_t* in_bf   = (bf16_t*)alloc((size_t)B_ * E_ * sizeof(bf16_t));
  bf16_t* x_bf    = (bf16_t*)alloc((size_t)B_ * E_ * sizeof(bf16_t));
  bf16_t* h_bf    = (bf16_t*)alloc((size_t)B_ * E_ * sizeof(bf16_t));
  float*  cbuf    = (float*) alloc((size_t)B_ * E_ * sizeof(float));
  float*  gates   = (float*) alloc((size_t)B_ * G_ * sizeof(float));
  float*  logits  = (float*) alloc((size_t)B_ * V_ * sizeof(float));
  unsigned char* lab = (unsigned char*)alloc((size_t)B_ * V_);
  float*  biasg   = (float*) alloc((size_t)G_ * sizeof(float));
  int*    yt      = (int*)   alloc((size_t)B_ * sizeof(int));
  float*  rloss   = (float*) alloc((size_t)B_ * sizeof(float));
  (void)ws_size; (void)in_sizes; (void)n_in; (void)out_size;

  const int T = 256;
  auto cdiv = [](size_t a, size_t b) { return (unsigned)((a + b - 1) / b); };

  // One-time per launch: bf16 weight/activation conversion + state init
  k_f2bf<<<cdiv((size_t)V_ * E_, T), T, 0, stream>>>(fc2_w, fc2w_bf, (size_t)V_ * E_);
  k_f2bf<<<cdiv((size_t)G_ * E_, T), T, 0, stream>>>(w_ih,  wih_bf,  (size_t)G_ * E_);
  k_f2bf<<<cdiv((size_t)G_ * E_, T), T, 0, stream>>>(w_hh,  whh_bf,  (size_t)G_ * E_);
  k_f2bf<<<cdiv((size_t)E_ * E_, T), T, 0, stream>>>(fc1_w, fc1w_bf, (size_t)E_ * E_);
  k_f2bf<<<cdiv((size_t)B_ * E_, T), T, 0, stream>>>(inputs, in_bf,  (size_t)B_ * E_);
  k_addbias<<<cdiv(G_, T), T, 0, stream>>>(b_ih, b_hh, biasg, G_);
  k_zero32<<<cdiv((size_t)B_ * E_, T), T, 0, stream>>>((unsigned*)cbuf, (size_t)B_ * E_);
  k_zero32<<<cdiv((size_t)B_ * E_ / 2, T), T, 0, stream>>>((unsigned*)h_bf, (size_t)B_ * E_ / 2);
  k_zero32<<<cdiv((size_t)B_ * V_, T), T, 0, stream>>>((unsigned*)d_out, (size_t)B_ * V_); // pred=0
  k_zero32<<<1, T, 0, stream>>>((unsigned*)rloss, (size_t)B_);
  k_copy32<<<cdiv((size_t)B_ * V_ / 4, T), T, 0, stream>>>(
      (const unsigned*)labels, (unsigned*)lab, (size_t)B_ * V_ / 4);

  // fc1: x = inputs @ fc1_w^T + fc1_b  (bf16 result for WMMA consumption)
  gemm_wmma<false><<<dim3(E_ / 16 / 4, 4), 128, 0, stream>>>(
      in_bf, fc1w_bf, nullptr, nullptr, fc1_b, nullptr, x_bf, E_, E_, E_ / 16);

  float* out_pred = (float*)d_out;
  float* out_loss = (float*)d_out + (size_t)B_ * V_;

  for (int t = 0; t < STEPS_; ++t) {
    // gates = x@w_ih^T + h@w_hh^T + (b_ih + b_hh)
    gemm_wmma<true><<<dim3(G_ / 16 / 4, 4), 128, 0, stream>>>(
        x_bf, wih_bf, h_bf, whh_bf, biasg, gates, nullptr, G_, E_, G_ / 16);
    // LSTM cell elementwise -> c, h(bf16)
    k_lstm<<<cdiv((size_t)B_ * E_, T), T, 0, stream>>>(gates, cbuf, h_bf);
    // logits = h @ fc2_w^T + fc2_b   (the big GEMM; fc2w_bf is L2 resident)
    gemm_wmma<false><<<dim3((V_ / 16 + 3) / 4, 4), 128, 0, stream>>>(
        h_bf, fc2w_bf, nullptr, nullptr, fc2_b, logits, nullptr, V_, E_, V_ / 16);
    // fused argmax + logsumexp + masked loss + label/pred update
    k_argmax_loss<<<B_, 256, 0, stream>>>(logits, lab, out_pred, yt, rloss, V_);
    // x_next = emb[yt] (bf16)
    k_gather<<<B_, 256, 0, stream>>>(emb, yt, x_bf);
  }

  k_finalize<<<1, 256, 0, stream>>>(rloss, out_loss);
}
